// MultiModalClinicalGAT_67757404062353
// MI455X (gfx1250) — matrix operand order repaired and checked by reading, not compiled, not run on hardware
//
#include <hip/hip_runtime.h>
#include <hip/hip_bf16.h>
#include <math.h>

// ---------------------------------------------------------------------------
// MultiModalClinicalGAT for MI455X (gfx1250): bf16 WMMA GEMMs + edge-parallel
// GAT attention with atomics. wave32 everywhere.
// GEMM: block tile 32x64, wave tile 16x16 x2 (register blocked), K-step 64
// => 4 x v_wmma_f32_16x16x32_bf16 per barrier pair.
// ---------------------------------------------------------------------------

typedef __attribute__((ext_vector_type(16))) __bf16         v16bf;
typedef __attribute__((ext_vector_type(16))) unsigned short v16us;
typedef __attribute__((ext_vector_type(8)))  float          v8f;

// round-to-nearest-even f32 -> bf16 (bit pattern)
__device__ __forceinline__ unsigned short f2bf(float f) {
    unsigned u = __float_as_uint(f);
    u += 0x7FFFu + ((u >> 16) & 1u);
    return (unsigned short)(u >> 16);
}

// order-preserving float<->uint mapping for atomicMax on floats
__device__ __forceinline__ unsigned mono(float f) {
    unsigned u = __float_as_uint(f);
    return (u & 0x80000000u) ? ~u : (u | 0x80000000u);
}
__device__ __forceinline__ float unmono(unsigned u) {
    return (u & 0x80000000u) ? __uint_as_float(u & 0x7FFFFFFFu)
                             : __uint_as_float(~u);
}

__device__ __forceinline__ void edge_nodes(const int* __restrict__ ei, int E,
                                           int eid, int& s, int& d) {
    if (eid < E) { s = ei[eid]; d = ei[E + eid]; }
    else         { s = d = eid - E; }          // self loops appended
}

// ---------------------------------------------------------------------------
// Tiled bf16 WMMA GEMM:  C[M,N] = act( concat(A1|A2)[M,K] @ B[K,N] + bias )
// block = 128 threads (4 waves). Block tile: 32 rows x 64 cols, K-step 64.
// Each wave: two 16x16 C tiles (row halves) sharing the B fragment.
// Requires K % 64 == 0, N % 64 == 0, split % 4 == 0.
// act: 0=none, 1=relu, 2=elu
// ---------------------------------------------------------------------------
__global__ __launch_bounds__(128)
void gemm_bf16_wmma(const float* __restrict__ A1, const float* __restrict__ A2,
                    int split, const float* __restrict__ B,
                    const float* __restrict__ bias, float* __restrict__ C,
                    int M, int N, int K, int act)
{
    __shared__ unsigned short As[32][68];   // 32 rows x 64 k (+pad)
    __shared__ unsigned short Bs[64][68];   // [col][k] transposed (+pad)

    const int tid  = threadIdx.x;
    const int wave = tid >> 5;
    const int lane = tid & 31;
    const int m0   = blockIdx.x * 32;
    const int n0   = blockIdx.y * 64;
    const int lda1 = split;
    const int lda2 = K - split;

    v8f acc0 = 0.0f;   // rows m0 .. m0+15
    v8f acc1 = 0.0f;   // rows m0+16 .. m0+31

    for (int kk = 0; kk < K; kk += 64) {
        // ---- stage A tile: 32 rows x 64 cols f32 -> bf16 (4 float4 / thread)
        {
            int r     = tid >> 2;             // 0..31
            int cbase = (tid & 3) * 16;       // 0,16,32,48
            int row   = m0 + r; if (row >= M) row = M - 1;
            #pragma unroll
            for (int i = 0; i < 4; ++i) {
                int c  = cbase + i * 4;
                int gc = kk + c;
                const float* p = (gc < split)
                    ? (A1 + (size_t)row * lda1 + gc)
                    : (A2 + (size_t)row * lda2 + (gc - split));
                float4 v = *(const float4*)p;
                As[r][c + 0] = f2bf(v.x); As[r][c + 1] = f2bf(v.y);
                As[r][c + 2] = f2bf(v.z); As[r][c + 3] = f2bf(v.w);
            }
            if (kk + 64 < K) {                // prefetch next A tile
                int gn = kk + 64 + cbase;
                const float* pn = (gn < split)
                    ? (A1 + (size_t)row * lda1 + gn)
                    : (A2 + (size_t)row * lda2 + (gn - split));
                __builtin_prefetch(pn, 0, 1);
            }
        }
        // ---- stage B tile: 64 k-rows x 64 cols -> Bs[col][k] (8 float4 / thread)
        #pragma unroll
        for (int i = 0; i < 8; ++i) {
            int lin = tid + 128 * i;          // 0..1023
            int k   = lin >> 4;               // 0..63
            int c   = (lin & 15) * 4;         // 0..60
            const float* p = B + (size_t)(kk + k) * N + n0 + c;
            float4 v = *(const float4*)p;
            Bs[c + 0][k] = f2bf(v.x); Bs[c + 1][k] = f2bf(v.y);
            Bs[c + 2][k] = f2bf(v.z); Bs[c + 3][k] = f2bf(v.w);
            if ((i & 3) == 0 && kk + 64 < K)
                __builtin_prefetch(p + (size_t)64 * N, 0, 1);
        }
        __syncthreads();

        // ---- 2 K-steps x 2 row-tiles = 4 WMMAs per iteration
        const int mrow = lane & 15;
        const int hi   = lane >> 4;           // half-wave select
        const int wcol = wave * 16 + mrow;
        #pragma unroll
        for (int ks = 0; ks < 2; ++ks) {
            const int kb0 = ks * 32;
            v16us au0, au1, bu;
            const int ka = kb0 + hi * 8;      // A: lanes0-15 K{0..7,16..23}; lanes16-31 K{8..15,24..31}
            #pragma unroll
            for (int j = 0; j < 4; ++j) {
                au0[2 * j]         = As[mrow][ka + 2 * j];
                au0[2 * j + 1]     = As[mrow][ka + 2 * j + 1];
                au0[8 + 2 * j]     = As[mrow][ka + 16 + 2 * j];
                au0[8 + 2 * j + 1] = As[mrow][ka + 16 + 2 * j + 1];
                au1[2 * j]         = As[16 + mrow][ka + 2 * j];
                au1[2 * j + 1]     = As[16 + mrow][ka + 2 * j + 1];
                au1[8 + 2 * j]     = As[16 + mrow][ka + 16 + 2 * j];
                au1[8 + 2 * j + 1] = As[16 + mrow][ka + 16 + 2 * j + 1];
            }
            const int kb = kb0 + hi * 8;      // B: lanes0-15 K{0..7,16..23}; lanes16-31 K{8..15,24..31}
            #pragma unroll
            for (int j = 0; j < 8; ++j) {
                bu[2 * j]     = Bs[wcol][kb + 2 * j + (j >= 4 ? 8 : 0)];
                bu[2 * j + 1] = Bs[wcol][kb + 2 * j + 1 + (j >= 4 ? 8 : 0)];
            }
            v16bf b = __builtin_bit_cast(v16bf, bu);
            acc0 = __builtin_amdgcn_wmma_f32_16x16x32_bf16(
                       false, __builtin_bit_cast(v16bf, au0), false, b,
                       (short)0, acc0, false, false);
            acc1 = __builtin_amdgcn_wmma_f32_16x16x32_bf16(
                       false, __builtin_bit_cast(v16bf, au1), false, b,
                       (short)0, acc1, false, false);
        }
        __syncthreads();
    }

    // ---- epilogue: lane holds col (lane&15), rows r + 8*(lane>>4)
    const int col = n0 + wave * 16 + (lane & 15);
    const int hi  = lane >> 4;
    float bv = bias ? bias[col] : 0.0f;
    #pragma unroll
    for (int r = 0; r < 8; ++r) {
        float v0 = acc0[r] + bv;
        float v1 = acc1[r] + bv;
        if (act == 1) {
            v0 = v0 > 0.0f ? v0 : 0.0f;
            v1 = v1 > 0.0f ? v1 : 0.0f;
        } else if (act == 2) {
            v0 = v0 > 0.0f ? v0 : (expf(v0) - 1.0f);
            v1 = v1 > 0.0f ? v1 : (expf(v1) - 1.0f);
        }
        int row0 = m0 + r + 8 * hi;
        int row1 = m0 + 16 + r + 8 * hi;
        if (row0 < M) C[(size_t)row0 * N + col] = v0;
        if (row1 < M) C[(size_t)row1 * N + col] = v1;
    }
}

// ---------------------------------------------------------------------------
// Attention helper kernels
// ---------------------------------------------------------------------------
__global__ void scores_kernel(const float* __restrict__ h,
                              const float* __restrict__ a_src,
                              const float* __restrict__ a_dst,
                              float* __restrict__ ssrc, float* __restrict__ sdst,
                              int n, int heads, int Cc)
{
    int i = blockIdx.x * blockDim.x + threadIdx.x;
    if (i >= n * heads) return;
    int node = i / heads, hh = i % heads;
    const float* hp = h + (size_t)node * heads * Cc + (size_t)hh * Cc;
    const float* as = a_src + hh * Cc;
    const float* ad = a_dst + hh * Cc;
    float s1 = 0.0f, s2 = 0.0f;
    for (int c = 0; c < Cc; ++c) { float v = hp[c]; s1 += v * as[c]; s2 += v * ad[c]; }
    ssrc[i] = s1; sdst[i] = s2;
}

__global__ void edge_max_kernel(const int* __restrict__ ei, int E, int Etot, int heads,
                                const float* __restrict__ ssrc,
                                const float* __restrict__ sdst,
                                unsigned* __restrict__ m)
{
    int i = blockIdx.x * blockDim.x + threadIdx.x;
    if (i >= Etot * heads) return;
    int eid = i / heads, hh = i % heads;
    int s, d; edge_nodes(ei, E, eid, s, d);
    float e = ssrc[s * heads + hh] + sdst[d * heads + hh];
    e = e > 0.0f ? e : 0.2f * e;                 // leaky_relu(0.2)
    atomicMax(m + d * heads + hh, mono(e));
}

__global__ void edge_exp_kernel(const int* __restrict__ ei, int E, int Etot, int heads,
                                const float* __restrict__ ssrc,
                                const float* __restrict__ sdst,
                                const unsigned* __restrict__ m,
                                float* __restrict__ exbuf, float* __restrict__ den)
{
    int i = blockIdx.x * blockDim.x + threadIdx.x;
    if (i >= Etot * heads) return;
    int eid = i / heads, hh = i % heads;
    int s, d; edge_nodes(ei, E, eid, s, d);
    float e = ssrc[s * heads + hh] + sdst[d * heads + hh];
    e = e > 0.0f ? e : 0.2f * e;
    float ex = expf(e - unmono(m[d * heads + hh]));
    exbuf[i] = ex;
    atomicAdd(den + d * heads + hh, ex);
}

// one 32-lane group per (edge, head); lanes stride the channel dim
__global__ __launch_bounds__(256)
void edge_scatter_kernel(const int* __restrict__ ei, int E, int Etot, int heads, int Cc,
                         const float* __restrict__ exbuf, const float* __restrict__ den,
                         const float* __restrict__ h, float* __restrict__ out)
{
    int g    = blockIdx.x * (blockDim.x >> 5) + (threadIdx.x >> 5);
    int lane = threadIdx.x & 31;
    if (g >= Etot * heads) return;
    int eid = g / heads, hh = g % heads;
    int s, d; edge_nodes(ei, E, eid, s, d);
    float alpha = exbuf[g] / (den[d * heads + hh] + 1e-16f);
    const float* hp = h + (size_t)s * heads * Cc + (size_t)hh * Cc;
    float*       op = out + (size_t)d * heads * Cc + (size_t)hh * Cc;
    for (int c = lane; c < Cc; c += 32)
        atomicAdd(op + c, hp[c] * alpha);
}

// thread per edge for the tiny C=4 layer
__global__ void edge_scatter_small(const int* __restrict__ ei, int E, int Etot,
                                   const float* __restrict__ exbuf,
                                   const float* __restrict__ den,
                                   const float* __restrict__ h,
                                   float* __restrict__ out, int Cc)
{
    int i = blockIdx.x * blockDim.x + threadIdx.x;
    if (i >= Etot) return;
    int s, d; edge_nodes(ei, E, i, s, d);
    float alpha = exbuf[i] / (den[d] + 1e-16f);
    for (int c = 0; c < Cc; ++c)
        atomicAdd(out + (size_t)d * Cc + c, h[(size_t)s * Cc + c] * alpha);
}

// in-place x[i] = act(x[i] + bias[i % F])
__global__ void bias_act_kernel(float* __restrict__ x, const float* __restrict__ bias,
                                long long n, int F, int act)
{
    long long i = (long long)blockIdx.x * blockDim.x + threadIdx.x;
    if (i >= n) return;
    float v = x[i] + bias[(int)(i % F)];
    if (act == 2) v = v > 0.0f ? v : (expf(v) - 1.0f);
    x[i] = v;
}

// fused: h2 = x2 @ W2 (256->4), plus per-node attention halves for layer 2
__global__ void lin2_scores_kernel(const float* __restrict__ x2,
                                   const float* __restrict__ W2,
                                   const float* __restrict__ a2s,
                                   const float* __restrict__ a2d,
                                   float* __restrict__ h2,
                                   float* __restrict__ s2src,
                                   float* __restrict__ s2dst, int n)
{
    int i = blockIdx.x * blockDim.x + threadIdx.x;
    if (i >= n) return;
    const float* xp = x2 + (size_t)i * 256;
    float o0 = 0, o1 = 0, o2 = 0, o3 = 0;
    for (int k = 0; k < 256; ++k) {
        float v = xp[k];
        const float* w = W2 + k * 4;
        o0 += v * w[0]; o1 += v * w[1]; o2 += v * w[2]; o3 += v * w[3];
    }
    h2[(size_t)i * 4 + 0] = o0; h2[(size_t)i * 4 + 1] = o1;
    h2[(size_t)i * 4 + 2] = o2; h2[(size_t)i * 4 + 3] = o3;
    s2src[i] = o0 * a2s[0] + o1 * a2s[1] + o2 * a2s[2] + o3 * a2s[3];
    s2dst[i] = o0 * a2d[0] + o1 * a2d[1] + o2 * a2d[2] + o3 * a2d[3];
}

// ---------------------------------------------------------------------------
extern "C" void kernel_launch(void* const* d_in, const int* in_sizes, int n_in,
                              void* d_out, int out_size, void* d_ws, size_t ws_size,
                              hipStream_t stream)
{
    const float* clinical = (const float*)d_in[0];   // [N,128]
    const float* mel      = (const float*)d_in[1];   // [N,2048] flattened
    const int*   ei       = (const int*)d_in[2];     // [2,E]
    const float* W_mel    = (const float*)d_in[3];   // [2048,128]
    const float* b_mel    = (const float*)d_in[4];
    const float* W_cat    = (const float*)d_in[5];   // [256,128]
    const float* b_cat    = (const float*)d_in[6];
    const float* W1       = (const float*)d_in[7];   // [128,256]
    const float* a1_src   = (const float*)d_in[8];   // [2,128]
    const float* a1_dst   = (const float*)d_in[9];
    const float* b1       = (const float*)d_in[10];  // [256]
    const float* W2       = (const float*)d_in[11];  // [256,4]
    const float* a2_src   = (const float*)d_in[12];  // [4]
    const float* a2_dst   = (const float*)d_in[13];
    const float* b2       = (const float*)d_in[14];  // [4]
    float* out = (float*)d_out;

    const int N    = in_sizes[0] / 128;
    const int E    = in_sizes[2] / 2;
    const int Etot = E + N;                          // self loops appended

    // ---- workspace layout (floats) -------------------------------------
    float* w = (float*)d_ws;
    size_t off = 0;
    float* melh = w + off; off += (size_t)N * 128;   // region reused below after GEMM2
    float* x1   = w + off; off += (size_t)N * 128;
    float* h1   = w + off; off += (size_t)N * 256;
    float* out1 = w + off; off += (size_t)N * 256;   // becomes x2 in place
    // small arrays carved out of melh (dead after GEMM2; all writes occur later)
    float*    s1src = melh;                     // N*2
    float*    s1dst = s1src + (size_t)N * 2;    // N*2
    unsigned* m1    = (unsigned*)(s1dst + (size_t)N * 2);  // N*2
    float*    den1  = (float*)m1 + (size_t)N * 2;          // N*2
    float*    ex1   = den1 + (size_t)N * 2;     // Etot*2
    float*    h2    = ex1 + (size_t)Etot * 2;   // N*4
    float*    s2src = h2 + (size_t)N * 4;       // N
    float*    s2dst = s2src + N;                // N
    unsigned* m2    = (unsigned*)(s2dst + N);   // N
    float*    den2  = (float*)m2 + N;           // N
    float*    ex2   = den2 + N;                 // Etot
    (void)ws_size; (void)n_in; (void)out_size;

    // ---- feature pipeline: three bf16-WMMA GEMMs -----------------------
    dim3 blk(128);
    // mel_h = relu(mel @ W_mel + b_mel)          [N,2048]x[2048,128]
    gemm_bf16_wmma<<<dim3((N + 31) / 32, 128 / 64), blk, 0, stream>>>(
        mel, mel, 2048, W_mel, b_mel, melh, N, 128, 2048, 1);
    // x1 = elu(concat(clinical, mel_h) @ W_cat + b_cat)   [N,256]x[256,128]
    gemm_bf16_wmma<<<dim3((N + 31) / 32, 128 / 64), blk, 0, stream>>>(
        clinical, melh, 128, W_cat, b_cat, x1, N, 128, 256, 2);
    // h1 = x1 @ W1                                [N,128]x[128,256]
    gemm_bf16_wmma<<<dim3((N + 31) / 32, 256 / 64), blk, 0, stream>>>(
        x1, x1, 128, W1, nullptr, h1, N, 256, 128, 0);

    // ---- zero accumulators (after GEMM2: melh region now reusable) -----
    hipMemsetAsync(m1,   0, (size_t)N * 2 * sizeof(unsigned), stream);
    hipMemsetAsync(den1, 0, (size_t)N * 2 * sizeof(float),    stream);
    hipMemsetAsync(out1, 0, (size_t)N * 256 * sizeof(float),  stream);
    hipMemsetAsync(m2,   0, (size_t)N * sizeof(unsigned),     stream);
    hipMemsetAsync(den2, 0, (size_t)N * sizeof(float),        stream);
    hipMemsetAsync(out,  0, (size_t)N * 4 * sizeof(float),    stream);

    // ---- GAT layer 1 (heads=2, C=128) ----------------------------------
    scores_kernel<<<(N * 2 + 255) / 256, 256, 0, stream>>>(
        h1, a1_src, a1_dst, s1src, s1dst, N, 2, 128);
    edge_max_kernel<<<(Etot * 2 + 255) / 256, 256, 0, stream>>>(
        ei, E, Etot, 2, s1src, s1dst, m1);
    edge_exp_kernel<<<(Etot * 2 + 255) / 256, 256, 0, stream>>>(
        ei, E, Etot, 2, s1src, s1dst, m1, ex1, den1);
    edge_scatter_kernel<<<(Etot * 2 + 7) / 8, 256, 0, stream>>>(
        ei, E, Etot, 2, 128, ex1, den1, h1, out1);
    // x2 = elu(out1 + b1), in place
    bias_act_kernel<<<(int)(((long long)N * 256 + 255) / 256), 256, 0, stream>>>(
        out1, b1, (long long)N * 256, 256, 2);

    // ---- GAT layer 2 (heads=1, C=4) ------------------------------------
    lin2_scores_kernel<<<(N + 255) / 256, 256, 0, stream>>>(
        out1, W2, a2_src, a2_dst, h2, s2src, s2dst, N);
    edge_max_kernel<<<(Etot + 255) / 256, 256, 0, stream>>>(
        ei, E, Etot, 1, s2src, s2dst, m2);
    edge_exp_kernel<<<(Etot + 255) / 256, 256, 0, stream>>>(
        ei, E, Etot, 1, s2src, s2dst, m2, ex2, den2);
    edge_scatter_small<<<(Etot + 255) / 256, 256, 0, stream>>>(
        ei, E, Etot, ex2, den2, h2, out, 4);
    // out += b2
    bias_act_kernel<<<(int)(((long long)N * 4 + 255) / 256), 256, 0, stream>>>(
        out, b2, (long long)N * 4, 4, 0);
}
